// TBCNN_73254962201169
// MI455X (gfx1250) — matrix operand order repaired and checked by reading, not compiled, not run on hardware
//
#include <hip/hip_runtime.h>
#include <hip/hip_bf16.h>
#include <math.h>

typedef __attribute__((ext_vector_type(16))) _Float16 v16h;
typedef __attribute__((ext_vector_type(8)))  _Float16 v8h;
typedef __attribute__((ext_vector_type(4)))  _Float16 v4h;
typedef __attribute__((ext_vector_type(8)))  float    v8f;
typedef __attribute__((ext_vector_type(4)))  float    v4f;

#define BB 32
#define NN 2048
#define CC 16
#define DD 128
#define HH 128
#define OO 128
#define K3 384      // 3*D
#define KSTEPS 12   // 384/32
#define NT 8        // 128/16 N-tiles
#define WF_PAD 392  // 384 + 8 halves: row stride 784B -> 4-bank skew per row

// ---- monotone float <-> uint encoding for atomic max pooling ----
__device__ __forceinline__ unsigned encf(float f) {
    unsigned u = __float_as_uint(f);
    return (u & 0x80000000u) ? ~u : (u | 0x80000000u);
}
__device__ __forceinline__ float decf(unsigned u) {
    return (u & 0x80000000u) ? __uint_as_float(u & 0x7FFFFFFFu) : __uint_as_float(~u);
}

// ---- kernel 0a: repack W1 (f32, 384x128 row-major) into f16 B-fragment order ----
// GEMM reads lane L's v16h for (kstep s, ntile t) at W1f[((s*8+t)*32 + L)*16 + e],
// expecting element W1[32s + 16*(L/16) + e][16t + (L%16)].
__global__ void k_prep_w1(const float* __restrict__ W1, _Float16* __restrict__ W1f) {
    int e = blockIdx.x * blockDim.x + threadIdx.x;
    if (e >= K3 * HH) return;
    int k = e / HH, n = e % HH;
    int s = k >> 5, kk = k & 31, g = kk >> 4, j = kk & 15;
    int t = n >> 4, ln = n & 15;
    int lane = g * 16 + ln;
    W1f[((s * NT + t) * 32 + lane) * 16 + j] = (_Float16)W1[e];
}

// ---- kernel 0b: init pooled max accumulators (enc identity = 0) ----
__global__ void k_init_pool(unsigned* __restrict__ pooled_enc) {
    int i = blockIdx.x * blockDim.x + threadIdx.x;
    if (i < BB * HH) pooled_enc[i] = 0u;
}

// ---- kernel 1: gather + conv-pool + GEMM(WMMA) + pre-activation max-pool ----
// NOTE: tanh is strictly monotone, so max_n tanh(x+b) = tanh(b + max_n x).
// We pool the raw accumulator max; bias + tanh are applied once in stage 2.
__global__ __launch_bounds__(128) void k_stage1(
    const int*   __restrict__ children,   // (B,N,C)
    const float* __restrict__ emb,        // (B,N,D)
    const _Float16* __restrict__ W1f,     // repacked
    unsigned*    __restrict__ pooled_enc) // (B,H) encoded max of pre-activation
{
    __shared__ _Float16 wf[4][16][WF_PAD];

    const int lane = threadIdx.x & 31;
    const int wv   = threadIdx.x >> 5;
    const int mtile = blockIdx.x * 4 + wv;        // 4096 tiles total
    const int b     = mtile >> 7;                 // 128 tiles per batch
    const int nbase = (mtile & 127) << 4;
    const int g  = lane >> 4;                     // half-wave group
    const int lr = lane & 15;
    const int d4 = lane * 4;                      // 4 D-elements per lane

    // ---------- Phase A: build wf tile (16 x 384) in LDS as f16 ----------
    for (int i = 0; i < 16; ++i) {
        const int node = nbase + i;
        const int rowBase = (b * NN + node) * DD;

        // self embedding -> cols [0,128)
        v4f ev = *(const v4f*)(emb + rowBase + d4);
        *(v4h*)&wf[wv][i][d4] =
            (v4h){(_Float16)ev.x, (_Float16)ev.y, (_Float16)ev.z, (_Float16)ev.w};

        // per-child conv weights (lanes 0..15 own child slots)
        int ch_l = 0;
        if (lane < 16) ch_l = children[(b * NN + node) * CC + lane];
        unsigned long long bal = __ballot(ch_l > 0);
        int num_sib = __popcll(bal & 0xFFFFull);
        float m  = (ch_l > 0) ? 1.0f : 0.0f;
        float wr = (num_sib == 1) ? 0.5f * m
                                  : m * (float)lane / (float)max(num_sib - 1, 1);
        float wl = m * (1.0f - wr);

        v4f racc = (v4f){0.f, 0.f, 0.f, 0.f};
        v4f lacc = (v4f){0.f, 0.f, 0.f, 0.f};
        #pragma unroll 4
        for (int c = 0; c < CC; ++c) {
            int ch = __shfl(ch_l, c);
            if (ch > 0) {  // lookup row 0 is zeroed in reference; masked anyway
                float cwr = __shfl(wr, c);
                float cwl = __shfl(wl, c);
                v4f cv = *(const v4f*)(emb + (b * NN + ch) * DD + d4);
                racc += cwr * cv;   // vector FMA -> v_pk_fma_f32 pairs
                lacc += cwl * cv;
            }
        }
        *(v4h*)&wf[wv][i][128 + d4] =
            (v4h){(_Float16)racc.x, (_Float16)racc.y, (_Float16)racc.z, (_Float16)racc.w};
        *(v4h*)&wf[wv][i][256 + d4] =
            (v4h){(_Float16)lacc.x, (_Float16)lacc.y, (_Float16)lacc.z, (_Float16)lacc.w};
    }
    __syncthreads();

    // ---------- Phase B: 16x384 @ 384x128 via WMMA f16 -> f32 ----------
    v8f acc[NT];
    #pragma unroll
    for (int t = 0; t < NT; ++t) acc[t] = (v8f){0.f,0.f,0.f,0.f,0.f,0.f,0.f,0.f};

    for (int s = 0; s < KSTEPS; ++s) {
        // A fragment: lane holds row lr, K = 32s+16g..+15
        v8h a0 = *(const v8h*)&wf[wv][lr][32 * s + 16 * g];
        v8h a1 = *(const v8h*)&wf[wv][lr][32 * s + 16 * g + 8];
        v16h a = __builtin_shufflevector(a0, a1,
                 0,1,2,3,4,5,6,7,8,9,10,11,12,13,14,15);
        #pragma unroll
        for (int t = 0; t < NT; ++t) {
            v16h bf = *(const v16h*)(W1f + ((s * NT + t) * 32 + lane) * 16);
            acc[t] = __builtin_amdgcn_wmma_f32_16x16x32_f16(
                         false, a, false, bf, (short)0, acc[t], false, false);
        }
    }

    // ---------- Epilogue: max over the 16 node rows (pre-activation) ----------
    #pragma unroll
    for (int t = 0; t < NT; ++t) {
        float mx = acc[t][0];
        #pragma unroll
        for (int v = 1; v < 8; ++v) mx = fmaxf(mx, acc[t][v]);
        mx = fmaxf(mx, __shfl_xor(mx, 16));    // combine M halves
        if (g == 0) atomicMax(pooled_enc + b * HH + (t * 16 + lr), encf(mx));
    }
}

// ---- kernel 2: out = tanh( tanh(max+b1) @ W2 + b2 ), (32x128)x(128x128) ----
__global__ void k_stage2(const unsigned* __restrict__ pooled_enc,
                         const float* __restrict__ b1,
                         const float* __restrict__ W2,
                         const float* __restrict__ b2,
                         float* __restrict__ outf) {
    int b = blockIdx.x, o = threadIdx.x;
    float s = b2[o];
    for (int h = 0; h < HH; ++h) {
        float pooled = tanhf(decf(pooled_enc[b * HH + h]) + b1[h]);
        s += pooled * W2[h * OO + o];
    }
    outf[b * OO + o] = tanhf(s);
}

// ---- kernel 3: 16 cosine similarities, one wave per pair ----
__global__ void k_cosine(const float* __restrict__ outf, float* __restrict__ res) {
    int lane = threadIdx.x & 31, w = threadIdx.x >> 5;  // 16 waves
    const float* v1 = outf + w * OO;
    const float* v2 = outf + (w + 16) * OO;
    float d = 0.f, s1 = 0.f, s2 = 0.f;
    #pragma unroll
    for (int j = 0; j < 4; ++j) {
        float a = v1[lane * 4 + j], c = v2[lane * 4 + j];
        d += a * c; s1 += a * a; s2 += c * c;
    }
    #pragma unroll
    for (int off = 16; off; off >>= 1) {
        d  += __shfl_xor(d,  off);
        s1 += __shfl_xor(s1, off);
        s2 += __shfl_xor(s2, off);
    }
    if (lane == 0) {
        float n1 = fmaxf(sqrtf(s1), 1e-8f);
        float n2 = fmaxf(sqrtf(s2), 1e-8f);
        res[w] = d / (n1 * n2);
    }
}

extern "C" void kernel_launch(void* const* d_in, const int* in_sizes, int n_in,
                              void* d_out, int out_size, void* d_ws, size_t ws_size,
                              hipStream_t stream) {
    const int*   children = (const int*)  d_in[0];  // (B,N,C)
    const float* nodeEmb  = (const float*)d_in[1];  // (B,N,D)
    const float* W1       = (const float*)d_in[2];  // (384,128)
    const float* b1       = (const float*)d_in[3];  // (128,)
    const float* W2       = (const float*)d_in[4];  // (128,128)
    const float* b2       = (const float*)d_in[5];  // (128,)
    float* out = (float*)d_out;                     // 16 floats

    // workspace layout
    _Float16* W1f        = (_Float16*)d_ws;                          // 98304 B
    unsigned* pooled_enc = (unsigned*)((char*)d_ws + 98304);         // 16384 B
    float*    outf       = (float*)   ((char*)d_ws + 98304 + 16384); // 16384 B

    k_prep_w1 <<<(K3 * HH + 255) / 256, 256, 0, stream>>>(W1, W1f);
    k_init_pool<<<(BB * HH + 255) / 256, 256, 0, stream>>>(pooled_enc);
    k_stage1  <<<(BB * NN / 16) / 4, 128, 0, stream>>>(children, nodeEmb, W1f, pooled_enc);
    k_stage2  <<<BB, HH, 0, stream>>>(pooled_enc, b1, W2, b2, outf);
    k_cosine  <<<1, 512, 0, stream>>>(outf, out);
}
